// SparseNodeEdgeAttentionLayer_65403761983980
// MI455X (gfx1250) — matrix operand order repaired and checked by reading, not compiled, not run on hardware
//
#include <hip/hip_runtime.h>
#include <hip/hip_bf16.h>

typedef __attribute__((ext_vector_type(16))) _Float16 v16h;
typedef __attribute__((ext_vector_type(8)))  float    v8f;

#define DIM_D 256   // input feature dim
#define DIM_A 128   // attention dim (q/k width)

// ---------------------------------------------------------------------------
// Kernel 1: zero diagA0 accumulator
// ---------------------------------------------------------------------------
__global__ void zero_f32_kernel(float* __restrict__ p, int n) {
    int i = blockIdx.x * blockDim.x + threadIdx.x;
    if (i < n) p[i] = 0.0f;
}

// ---------------------------------------------------------------------------
// Kernel 2: projection GEMM  out[N,128] = x[N,256] @ W^T[256,128] + bias
// via v_wmma_f32_16x16x32_f16 (f16 inputs, f32 accumulate).
// Block = 256 threads = 8 waves; block b owns rows [16b,16b+16); wave w owns
// output cols [16w,16w+16). K-loop: 8 x (16x16x32) WMMAs, fully unrolled.
// Requires N % 16 == 0 (N = 50000 -> 3125 row blocks exactly).
// ---------------------------------------------------------------------------
__global__ __launch_bounds__(256) void proj_wmma_kernel(
    const float* __restrict__ x, const float* __restrict__ W,
    const float* __restrict__ bias, float* __restrict__ out)
{
    const int wave = threadIdx.x >> 5;          // 0..7 : column tile
    const int lane = threadIdx.x & 31;
    const int l16  = lane & 15;
    const int hi   = lane >> 4;                 // lane half

    const int m = blockIdx.x * 16 + l16;        // A-matrix row for this lane
    const int n = wave * 16 + l16;              // B-matrix col for this lane

    const float* __restrict__ xrow = x + (size_t)m * DIM_D;
    const float* __restrict__ wrow = W + (size_t)n * DIM_D;

    v8f acc = {};

    #pragma unroll
    for (int kk = 0; kk < DIM_D; kk += 32) {
        // A fragment: 16x32 f16, lane = M, per ISA 7.12.2:
        //   elems 0..7  -> K = kk + 8*hi + [0..7]
        //   elems 8..15 -> K = kk + 16 + 8*hi + [0..7]
        v16h a, b;
        const float* xa = xrow + kk + 8 * hi;
        #pragma unroll
        for (int j = 0; j < 8; ++j) a[j]     = (_Float16)xa[j];
        #pragma unroll
        for (int j = 0; j < 8; ++j) a[8 + j] = (_Float16)xa[16 + j];

        // B fragment: 32x16 f16, lane = N, same K order as A:
        //   elems 0..7 -> K = kk + 8*hi + [0..7], elems 8..15 -> K = kk + 16 + 8*hi + [0..7]
        // B[k][n] = W[n][k]  (out = x @ W^T)
        const float* wb = wrow + kk + 8 * hi;
        #pragma unroll
        for (int i = 0; i < 16; ++i) b[i] = (_Float16)wb[i < 8 ? i : 8 + i];

        acc = __builtin_amdgcn_wmma_f32_16x16x32_f16(
            /*neg_a=*/false, a, /*neg_b=*/false, b,
            /*c_mod=*/(short)0, acc, /*reuse_a=*/false, /*reuse_b=*/false);
    }

    // C/D layout: VGPR r holds row M = r + 8*hi, col N = lane&15
    const float bv = bias[n];
    float* __restrict__ orow =
        out + ((size_t)blockIdx.x * 16 + 8 * hi) * DIM_A + n;
    #pragma unroll
    for (int r = 0; r < 8; ++r)
        orow[(size_t)r * DIM_A] = acc[r] + bv;
}

// ---------------------------------------------------------------------------
// Kernel 3: per-edge symmetric attention score + exp + scatter-add.
// One wave32 per edge; lane owns 4 of the 128 dims (float4 loads).
// pre = (0.5/H) * (q[src].k[dst] + q[dst].k[src]),  H = 8  -> scale 0.0625
// ---------------------------------------------------------------------------
__global__ __launch_bounds__(256) void edge_attn_kernel(
    const float* __restrict__ q, const float* __restrict__ k,
    const int* __restrict__ edge_index,   // [2,E] flat: src then dst
    const int* __restrict__ d0_row1,      // d0_index[1], length 2E
    float* __restrict__ diagA0, float* __restrict__ diagA1, int E)
{
    const int e = blockIdx.x * 8 + (threadIdx.x >> 5);
    if (e >= E) return;
    const int lane = threadIdx.x & 31;

    const int src = edge_index[e];
    const int dst = edge_index[E + e];

    const float4* __restrict__ qi = (const float4*)(q + (size_t)src * DIM_A);
    const float4* __restrict__ kj = (const float4*)(k + (size_t)dst * DIM_A);
    const float4* __restrict__ qj = (const float4*)(q + (size_t)dst * DIM_A);
    const float4* __restrict__ ki = (const float4*)(k + (size_t)src * DIM_A);

    const float4 a = qi[lane];
    const float4 b = kj[lane];
    const float4 c = qj[lane];
    const float4 d = ki[lane];

    float r = a.x * b.x + a.y * b.y + a.z * b.z + a.w * b.w
            + c.x * d.x + c.y * d.y + c.z * d.z + c.w * d.w;

    // wave32 butterfly reduction
    #pragma unroll
    for (int off = 16; off >= 1; off >>= 1)
        r += __shfl_xor(r, off, 32);

    if (lane == 0) {
        const float s = __expf(r * 0.0625f);
        diagA1[e] = s;
        atomicAdd(diagA0 + d0_row1[2 * e],     s);
        atomicAdd(diagA0 + d0_row1[2 * e + 1], s);
    }
}

// ---------------------------------------------------------------------------
// Launch: zero diagA0 -> project q -> project k -> edge scores + scatter
// Workspace: q [N*128 f32] then k [N*128 f32]  (~51.2 MB total)
// d_out: diagA0 [N] then diagA1 [E]
// ---------------------------------------------------------------------------
extern "C" void kernel_launch(void* const* d_in, const int* in_sizes, int n_in,
                              void* d_out, int out_size, void* d_ws, size_t ws_size,
                              hipStream_t stream) {
    const float* x  = (const float*)d_in[0];
    const float* Wq = (const float*)d_in[1];
    const float* bq = (const float*)d_in[2];
    const float* Wk = (const float*)d_in[3];
    const float* bk = (const float*)d_in[4];
    const int* edge_index = (const int*)d_in[5];
    const int* d0_index   = (const int*)d_in[6];

    const int N = in_sizes[0] / DIM_D;     // 50000
    const int E = in_sizes[5] / 2;         // 800000

    float* q = (float*)d_ws;
    float* k = q + (size_t)N * DIM_A;
    float* diagA0 = (float*)d_out;
    float* diagA1 = diagA0 + N;

    zero_f32_kernel<<<(N + 255) / 256, 256, 0, stream>>>(diagA0, N);

    const int row_blocks = N / 16;         // N % 16 == 0 for this problem
    proj_wmma_kernel<<<row_blocks, 256, 0, stream>>>(x, Wq, bq, q);
    proj_wmma_kernel<<<row_blocks, 256, 0, stream>>>(x, Wk, bk, k);

    edge_attn_kernel<<<(E + 7) / 8, 256, 0, stream>>>(
        q, k, edge_index, d0_index + 2 * (size_t)E, diagA0, diagA1, E);
}
